// CoarseMatching_63685775065401
// MI455X (gfx1250) — matrix operand order repaired and checked by reading, not compiled, not run on hardware
//
#include <hip/hip_runtime.h>

typedef __attribute__((ext_vector_type(16))) _Float16 v16h;
typedef __attribute__((ext_vector_type(8)))  float    v8f;
typedef __attribute__((ext_vector_type(4)))  float    f32x4;

union HFrag { v16h h; f32x4 q[2]; };

#define NPIX 6400   // 80*80
#define CDIM 256
#define WIMG 80
#define INV_T 10.0f // 1/temperature
#define THRESH 0.2f

// ---------------------------------------------------------------------------
// K1: L2-normalize features along C; output row-major [NPIX][CDIM] f16.
// Input layout is (C, H*W): element F[c*NPIX + n].
// ---------------------------------------------------------------------------
__global__ __launch_bounds__(256) void k_normalize(const float* __restrict__ f0,
                                                   const float* __restrict__ f1,
                                                   _Float16* __restrict__ o0,
                                                   _Float16* __restrict__ o1) {
  const float* __restrict__ F = blockIdx.y ? f1 : f0;
  _Float16* __restrict__ O    = blockIdx.y ? o1 : o0;
  const int n = blockIdx.x * 256 + threadIdx.x;
  float ss = 0.f;
#pragma unroll 8
  for (int c = 0; c < CDIM; ++c) { float v = F[c * NPIX + n]; ss = fmaf(v, v, ss); }
  const float inv = 1.0f / fmaxf(sqrtf(ss), 1e-12f);
#pragma unroll 8
  for (int c = 0; c < CDIM; ++c) O[(size_t)n * CDIM + c] = (_Float16)(F[c * NPIX + n] * inv);
}

// ---------------------------------------------------------------------------
// K2: S = (A * B^T) * INV_T via v_wmma_f32_16x16x32_f16.
// A,B row-major [NPIX][CDIM] f16. Wave computes 64x32 output; WG (8 waves)
// computes 128x128; grid 50x50.
// A-frag (16x32): lane l<16 holds row M=l, K chunks [0..7] and [16..23];
//                 lane l+16 holds K [8..15] and [24..31]  (two b128 loads).
// B-frag (32x16): lane l<16 holds col N=l = row n of B-source, K [0..7] and [16..23];
//                 lane l+16 holds K [8..15] and [24..31]  (two b128 loads).
// C-frag 16x16 f32: VGPR r, lane(l,ph) -> element [M = r + 8*ph][N = l].
// ---------------------------------------------------------------------------
__global__ __launch_bounds__(256) void k_gemm(const _Float16* __restrict__ A,
                                              const _Float16* __restrict__ B,
                                              float* __restrict__ S) {
  const int lane = threadIdx.x & 31;
  const int wave = threadIdx.x >> 5;
  const int l    = lane & 15;
  const int ph   = lane >> 4;         // half of wave
  const int wm   = wave >> 2;         // 0..1
  const int wn   = wave & 3;          // 0..3
  const int mBase = blockIdx.y * 128 + wm * 64;
  const int nBase = blockIdx.x * 128 + wn * 32;

  v8f acc[4][2] = {};

  for (int k0 = 0; k0 < CDIM; k0 += 32) {
    HFrag a[4];
#pragma unroll
    for (int t = 0; t < 4; ++t) {
      const _Float16* ap = A + (size_t)(mBase + t * 16 + l) * CDIM + k0 + 8 * ph;
      a[t].q[0] = *(const f32x4*)(ap);
      a[t].q[1] = *(const f32x4*)(ap + 16);
      if (k0 + 32 < CDIM) __builtin_prefetch(ap + 32, 0, 1); // global_prefetch_b8
    }
    HFrag b[2];
#pragma unroll
    for (int u = 0; u < 2; ++u) {
      const _Float16* bp = B + (size_t)(nBase + u * 16 + l) * CDIM + k0 + 8 * ph;
      b[u].q[0] = *(const f32x4*)(bp);
      b[u].q[1] = *(const f32x4*)(bp + 16);
    }
#pragma unroll
    for (int t = 0; t < 4; ++t)
#pragma unroll
      for (int u = 0; u < 2; ++u)
        acc[t][u] = __builtin_amdgcn_wmma_f32_16x16x32_f16(
            false, a[t].h, false, b[u].h, (short)0, acc[t][u], false, false);
  }

#pragma unroll
  for (int t = 0; t < 4; ++t)
#pragma unroll
    for (int u = 0; u < 2; ++u) {
      const int m0 = mBase + t * 16 + 8 * ph;
      const int n0 = nBase + u * 16 + l;
#pragma unroll
      for (int r = 0; r < 8; ++r)
        S[(size_t)(m0 + r) * NPIX + n0] = acc[t][u][r] * INV_T;
    }
}

// ---------------------------------------------------------------------------
// K3: per-row max and sum(exp(s-max)); one wave per row, 8 rows per block.
// ---------------------------------------------------------------------------
__global__ __launch_bounds__(256) void k_rowstats(const float* __restrict__ S,
                                                  float* __restrict__ rmax,
                                                  float* __restrict__ rsum) {
  const int lane = threadIdx.x & 31;
  const int row  = blockIdx.x * 8 + (threadIdx.x >> 5);
  const float* __restrict__ p = S + (size_t)row * NPIX;
  float m = -3.4e38f;
  for (int j = lane; j < NPIX; j += 32) m = fmaxf(m, p[j]);
#pragma unroll
  for (int off = 16; off; off >>= 1) m = fmaxf(m, __shfl_xor(m, off, 32));
  float s = 0.f;
  for (int j = lane; j < NPIX; j += 32) s += __expf(p[j] - m);
#pragma unroll
  for (int off = 16; off; off >>= 1) s += __shfl_xor(s, off, 32);
  if (lane == 0) { rmax[row] = m; rsum[row] = s; }
}

// ---------------------------------------------------------------------------
// K4: per-column max and sum(exp(s-max)); one thread per column (coalesced).
// ---------------------------------------------------------------------------
__global__ __launch_bounds__(256) void k_colstats(const float* __restrict__ S,
                                                  float* __restrict__ cmax,
                                                  float* __restrict__ csum) {
  const int j = blockIdx.x * 256 + threadIdx.x;
  float m = -3.4e38f;
  for (int i = 0; i < NPIX; ++i) m = fmaxf(m, S[(size_t)i * NPIX + j]);
  float s = 0.f;
  for (int i = 0; i < NPIX; ++i) s += __expf(S[(size_t)i * NPIX + j] - m);
  cmax[j] = m;
  csum[j] = s;
}

// ---------------------------------------------------------------------------
// K5: in-place P = softmax_row * softmax_col, float4-vectorized.
// ---------------------------------------------------------------------------
__global__ __launch_bounds__(256) void k_pwrite(float* __restrict__ P,
                                                const float* __restrict__ rmax,
                                                const float* __restrict__ rsum,
                                                const float* __restrict__ cmax,
                                                const float* __restrict__ csum) {
  const size_t g = ((size_t)blockIdx.x * 256 + threadIdx.x) * 4;
  const int i = (int)(g / NPIX);
  const int j = (int)(g % NPIX);
  const float rm = rmax[i];
  const float ri = 1.0f / rsum[i];
  f32x4 s = *(const f32x4*)(P + g);
  f32x4 o;
#pragma unroll
  for (int k = 0; k < 4; ++k)
    o[k] = __expf(s[k] - rm) * ri * __expf(s[k] - cmax[j + k]) / csum[j + k];
  *(f32x4*)(P + g) = o;
}

// ---------------------------------------------------------------------------
// K6: row max + argmax of P; one wave per row (first-occurrence tie-break).
// ---------------------------------------------------------------------------
__global__ __launch_bounds__(256) void k_rowargmax(const float* __restrict__ P,
                                                   float* __restrict__ conf,
                                                   int* __restrict__ idx0) {
  const int lane = threadIdx.x & 31;
  const int row  = blockIdx.x * 8 + (threadIdx.x >> 5);
  const float* __restrict__ p = P + (size_t)row * NPIX;
  float m = -1.0f;
  int bi = 0;
  for (int j = lane; j < NPIX; j += 32) {
    float v = p[j];
    if (v > m) { m = v; bi = j; }
  }
#pragma unroll
  for (int off = 16; off; off >>= 1) {
    float om = __shfl_xor(m, off, 32);
    int   oi = __shfl_xor(bi, off, 32);
    if (om > m || (om == m && oi < bi)) { m = om; bi = oi; }
  }
  if (lane == 0) { conf[row] = m; idx0[row] = bi; }
}

// ---------------------------------------------------------------------------
// K7: column argmax of P; one thread per column.
// ---------------------------------------------------------------------------
__global__ __launch_bounds__(256) void k_colargmax(const float* __restrict__ P,
                                                   int* __restrict__ idx1) {
  const int j = blockIdx.x * 256 + threadIdx.x;
  float m = -1.0f;
  int bi = 0;
  for (int i = 0; i < NPIX; ++i) {
    float v = P[(size_t)i * NPIX + j];
    if (v > m) { m = v; bi = i; }
  }
  idx1[j] = bi;
}

// ---------------------------------------------------------------------------
// K8: mutual-NN + threshold; write mkpts0/mkpts1/mconf/valid.
// ---------------------------------------------------------------------------
__global__ __launch_bounds__(256) void k_match(const float* __restrict__ conf,
                                               const int* __restrict__ idx0,
                                               const int* __restrict__ idx1,
                                               float* __restrict__ mk0,
                                               float* __restrict__ mk1,
                                               float* __restrict__ mcf,
                                               float* __restrict__ vld) {
  const int i  = blockIdx.x * 256 + threadIdx.x;
  const int i0 = idx0[i];
  const float c = conf[i];
  const bool valid = (idx1[i0] == i) && (c > THRESH);
  const float vf = valid ? 1.0f : 0.0f;
  mk0[i * 2 + 0] = (float)(i % WIMG) * vf;
  mk0[i * 2 + 1] = (float)(i / WIMG) * vf;
  mk1[i * 2 + 0] = (float)(i0 % WIMG) * vf;
  mk1[i * 2 + 1] = (float)(i0 / WIMG) * vf;
  mcf[i] = valid ? c : 0.0f;
  vld[i] = vf;
}

// ---------------------------------------------------------------------------
extern "C" void kernel_launch(void* const* d_in, const int* in_sizes, int n_in,
                              void* d_out, int out_size, void* d_ws, size_t ws_size,
                              hipStream_t stream) {
  const float* f0 = (const float*)d_in[0];
  const float* f1 = (const float*)d_in[1];

  // Workspace layout (~6.73 MB)
  _Float16* f0h = (_Float16*)d_ws;
  _Float16* f1h = f0h + (size_t)NPIX * CDIM;
  float* rmax = (float*)(f1h + (size_t)NPIX * CDIM);
  float* rsum = rmax + NPIX;
  float* cmax = rsum + NPIX;
  float* csum = cmax + NPIX;
  float* conf = csum + NPIX;
  int*   idx0 = (int*)(conf + NPIX);
  int*   idx1 = idx0 + NPIX;

  // Output layout: P, mkpts0, mkpts1, mconf, valid (all f32)
  float* P   = (float*)d_out;
  float* mk0 = P + (size_t)NPIX * NPIX;
  float* mk1 = mk0 + 2 * NPIX;
  float* mcf = mk1 + 2 * NPIX;
  float* vld = mcf + NPIX;

  k_normalize<<<dim3(NPIX / 256, 2), 256, 0, stream>>>(f0, f1, f0h, f1h);
  k_gemm     <<<dim3(NPIX / 128, NPIX / 128), 256, 0, stream>>>(f0h, f1h, P);
  k_rowstats <<<NPIX / 8, 256, 0, stream>>>(P, rmax, rsum);
  k_colstats <<<NPIX / 256, 256, 0, stream>>>(P, cmax, csum);
  k_pwrite   <<<(unsigned)(((size_t)NPIX * NPIX) / (256 * 4)), 256, 0, stream>>>(P, rmax, rsum, cmax, csum);
  k_rowargmax<<<NPIX / 8, 256, 0, stream>>>(P, conf, idx0);
  k_colargmax<<<NPIX / 256, 256, 0, stream>>>(P, idx1);
  k_match    <<<NPIX / 256, 256, 0, stream>>>(conf, idx0, idx1, mk0, mk1, mcf, vld);
}
